// DecoderBlock_41369124995489
// MI455X (gfx1250) — matrix operand order repaired and checked by reading, not compiled, not run on hardware
//
#include <hip/hip_runtime.h>

#define N_COARSE 16000
#define N_FINE   128000
#define N_KEEP   100000
#define C_IN     128
#define C_OUT    64
#define EPS      1e-5f

typedef __attribute__((ext_vector_type(16))) __bf16 v16bf;
typedef __attribute__((ext_vector_type(8)))  float  v8f;

__device__ __forceinline__ unsigned short f2bf(float f) {
  __bf16 h = (__bf16)f;
  return __builtin_bit_cast(unsigned short, h);
}
__device__ __forceinline__ __bf16 bfld(unsigned short u) {
  return __builtin_bit_cast(__bf16, u);
}

__device__ __forceinline__ v8f wmma_bf16(v16bf a, v16bf b, v8f c) {
  // 8 args: (neg_a, A, neg_b, B, c_mod, C, reuse_a, reuse_b)
  return __builtin_amdgcn_wmma_f32_16x16x32_bf16(false, a, false, b, (short)0, c,
                                                 false, false);
}

// ---------------------------------------------------------------------------
// Weight prep: fp32 -> bf16, transposed to [.., o, c] so each lane's B
// fragment is 16 contiguous bf16 (one 32B load).
//   wupT  layout: [j=512][c=128], j = k*64 + o   (from Wup[8][128][64])
//   wresT layout: [g=6][k=27][o=64][c=64]        (from Wres[3][2][27][64][64])
// ---------------------------------------------------------------------------
__global__ void cvt_weights_kernel(const float* __restrict__ Wup,
                                   const float* __restrict__ Wres,
                                   unsigned short* __restrict__ wupT,
                                   unsigned short* __restrict__ wresT) {
  const int NUP  = 8 * C_IN * C_OUT;        // 65536
  const int NRES = 6 * 27 * C_OUT * C_OUT;  // 663552
  int e = blockIdx.x * blockDim.x + threadIdx.x;
  if (e < NUP) {
    int j = e >> 7, c = e & 127;
    int k = j >> 6, o = j & 63;
    wupT[e] = f2bf(Wup[(k * C_IN + c) * C_OUT + o]);
  } else if (e < NUP + NRES) {
    int e2 = e - NUP;
    int c  = e2 & 63;
    int o  = (e2 >> 6) & 63;
    int gk = e2 >> 12;  // g*27 + k
    wresT[e2] = f2bf(Wres[(gk * C_OUT + c) * C_OUT + o]);
  }
}

// Zero the pad row (index N_KEEP) of the bf16 activation buffers so the
// spconv gather needs no bounds branch. Re-run every call (graph-safe).
__global__ void zero_pad_kernel(unsigned short* __restrict__ dec_bf,
                                unsigned short* __restrict__ h_bf) {
  int t = threadIdx.x;
  if (t < C_OUT) {
    dec_bf[(size_t)N_KEEP * C_OUT + t] = 0;
    h_bf[(size_t)N_KEEP * C_OUT + t]   = 0;
  }
}

// ---------------------------------------------------------------------------
// up[n*8+k][o] = sum_c x[n][c] * Wup[k][c][o]
// Block: 256 threads (8 waves), 16-row M tile; grid (1000, 4).
// ---------------------------------------------------------------------------
__global__ void __launch_bounds__(256)
gemm_up_kernel(const float* __restrict__ X, const unsigned short* __restrict__ WT,
               float* __restrict__ up) {
  __shared__ unsigned short As[16][C_IN + 8];
  const int row0 = blockIdx.x * 16;
  const int tid  = threadIdx.x;
  {  // cooperative fp32 -> bf16 stage of 16x128 A tile
    int r  = tid >> 4;         // 0..15
    int cb = (tid & 15) * 8;   // 0..120
    const float* p = X + (size_t)(row0 + r) * C_IN + cb;
#pragma unroll
    for (int j = 0; j < 8; ++j) As[r][cb + j] = f2bf(p[j]);
  }
  __syncthreads();

  const int wave = tid >> 5, lane = tid & 31;
  const int ct = blockIdx.y * 8 + wave;  // 0..31
  const int j0 = ct * 16;
  const int n  = lane & 15;   // A row M and B col N for this lane
  const int kh = lane >> 4;   // K-half select (this code's contiguous k = 16*kh + s in A and B)

  v8f acc = {};
#pragma unroll
  for (int ch = 0; ch < 4; ++ch) {
    const int c0 = ch * 32;
    v16bf a, b;
#pragma unroll
    for (int j = 0; j < 8; ++j) {
      a[j]     = bfld(As[n][c0 + kh * 16 + j]);
      a[8 + j] = bfld(As[n][c0 + kh * 16 + 8 + j]);
    }
    b = *reinterpret_cast<const v16bf*>(WT + (size_t)(j0 + n) * C_IN + c0 + kh * 16);
    acc = wmma_bf16(a, b, acc);
  }

  const int j = j0 + n;
  const int k = j >> 6, o = j & 63;
  const int mBase = kh * 8;
#pragma unroll
  for (int r = 0; r < 8; ++r) {
    int nrow = row0 + mBase + r;
    up[((size_t)nrow * 8 + k) * C_OUT + o] = acc[r];
  }
}

// ---------------------------------------------------------------------------
// Sparse conv: out[n][o] = sum_{k<27} sum_c xp[nbr[k][n]][c] * W[k][c][o]
// srcbf: (N_KEEP+1) x 64 bf16 activations (row N_KEEP is zeros).
// Block: 256 threads (8 waves) own a 32-row M tile; wave w -> M subtile
// (w>>2), N subtile (w&3)*16. Per tap: the 32x64 bf16 A tile (32 rows x 128B)
// is gathered straight into LDS with GLOBAL_LOAD_ASYNC_TO_LDS_B128 (one b128
// per thread), double-buffered across taps and synchronized with
// s_wait_asynccnt + workgroup barrier so gathers overlap the WMMAs.
// ---------------------------------------------------------------------------
__global__ void __launch_bounds__(256)
spconv_wmma_kernel(const unsigned short* __restrict__ srcbf,
                   const unsigned short* __restrict__ WT,
                   const int* __restrict__ nbr, float* __restrict__ out) {
  __shared__ unsigned short As[2][32][80];  // 80 = 64 + pad (16B-aligned rows)
  const int row0 = blockIdx.x * 32;
  const int tid  = threadIdx.x;
  const int lane = tid & 31;
  const int wave = tid >> 5;
  const int mt   = (wave >> 2) * 16;  // M subtile base row in tile
  const int o0   = (wave & 3) * 16;   // N subtile base col
  const int n    = lane & 15;
  const int kh   = lane >> 4;
  const int ldr  = tid >> 3;          // 0..31 (gather row in tile)
  const int ldc  = (tid & 7) * 8;     // bf16 element offset, 16B segments

  auto issue = [&](int k, int buf) {
    int idx = nbr[k * N_KEEP + row0 + ldr];  // may be N_KEEP -> zero pad row
    const unsigned short* gp = srcbf + (size_t)idx * C_OUT + ldc;
    unsigned lo = (unsigned)(size_t)&As[buf][ldr][ldc];
    asm volatile("global_load_async_to_lds_b128 %0, %1, off"
                 :: "v"(lo), "v"(gp)
                 : "memory");
  };

  issue(0, 0);
  issue(1, 1);

  v8f acc = {};
  for (int k = 0; k < 27; ++k) {
    // Async loads complete in order: <=1 outstanding means tap k has landed.
    if (k + 1 < 27)
      asm volatile("s_wait_asynccnt 0x1" ::: "memory");
    else
      asm volatile("s_wait_asynccnt 0x0" ::: "memory");
    __syncthreads();  // all waves' gathers for tap k visible

    const unsigned short(*Ab)[80] = As[k & 1];
    const unsigned short* wk = WT + (size_t)k * C_OUT * C_OUT;
#pragma unroll
    for (int ch = 0; ch < 2; ++ch) {
      const int c0 = ch * 32;
      v16bf a, b;
#pragma unroll
      for (int j = 0; j < 8; ++j) {
        a[j]     = bfld(Ab[mt + n][c0 + kh * 16 + j]);
        a[8 + j] = bfld(Ab[mt + n][c0 + kh * 16 + 8 + j]);
      }
      b = *reinterpret_cast<const v16bf*>(wk + (size_t)(o0 + n) * C_OUT + c0 + kh * 16);
      acc = wmma_bf16(a, b, acc);
    }

    __syncthreads();  // all waves done reading As[k&1]
    if (k + 2 < 27) issue(k + 2, k & 1);
  }

  const int mBase = kh * 8;
#pragma unroll
  for (int r = 0; r < 8; ++r) {
    out[(size_t)(row0 + mt + mBase + r) * C_OUT + o0 + n] = acc[r];
  }
}

// ---------------------------------------------------------------------------
// Per-channel sum / sum-of-squares: one block per channel, deterministic tree
// reduction (no atomics, no pre-zeroing). stats[c]=sum, stats[C+c]=sumsq.
// ---------------------------------------------------------------------------
__global__ void __launch_bounds__(256)
col_stats_kernel(const float* __restrict__ X, int R, int C, float* __restrict__ stats) {
  const int c   = blockIdx.x;
  const int tid = threadIdx.x;
  float s = 0.f, s2 = 0.f;
  for (int r = tid; r < R; r += 256) {
    float v = X[(size_t)r * C + c];
    s += v;
    s2 += v * v;
  }
  __shared__ float sh[512];
  sh[tid] = s;
  sh[256 + tid] = s2;
  __syncthreads();
  for (int st = 128; st > 0; st >>= 1) {
    if (tid < st) {
      sh[tid] += sh[tid + st];
      sh[256 + tid] += sh[256 + tid + st];
    }
    __syncthreads();
  }
  if (tid == 0) {
    stats[c]     = sh[0];
    stats[C + c] = sh[256];
  }
}

// y[i][c<64] = relu(bn(up[keep_idx[i]][c])), y[i][64..66] = coords_f[i]
__global__ void build_y_kernel(const float* __restrict__ up, const float* __restrict__ stats,
                               const float* __restrict__ g, const float* __restrict__ b,
                               const int* __restrict__ keep_idx,
                               const float* __restrict__ coords, float* __restrict__ y) {
  int e = blockIdx.x * blockDim.x + threadIdx.x;
  if (e >= N_KEEP * 67) return;
  int i = e / 67, c = e % 67;
  float v;
  if (c < C_OUT) {
    float m   = stats[c] * (1.f / N_FINE);
    float var = stats[C_OUT + c] * (1.f / N_FINE) - m * m;
    float x   = up[(size_t)keep_idx[i] * C_OUT + c];
    v = (x - m) * rsqrtf(var + EPS) * g[c] + b[c];
    v = v > 0.f ? v : 0.f;
  } else {
    v = coords[i * 3 + (c - C_OUT)];
  }
  y[e] = v;
}

// dec[i][o] = sum_c bn(y[i][c]) * Wr[c][o] + br[o] + shortcut[i][o]
// BN folded into per-channel scale/shift in LDS. Also emits bf16 copy.
__global__ void __launch_bounds__(256)
dec0_gemm_kernel(const float* __restrict__ y, const float* __restrict__ stats,
                 const float* __restrict__ g_r, const float* __restrict__ b_r,
                 const float* __restrict__ Wr, const float* __restrict__ br,
                 const float* __restrict__ shortcut, float* __restrict__ dec,
                 unsigned short* __restrict__ dec_bf) {
  __shared__ float sc[67], tc[67];
  __shared__ float yr[4][68];
  const int tid = threadIdx.x;
  const int i0  = blockIdx.x * 4;
  if (tid < 67) {
    float m   = stats[tid] * (1.f / N_KEEP);
    float var = stats[67 + tid] * (1.f / N_KEEP) - m * m;
    float s   = g_r[tid] * rsqrtf(var + EPS);
    sc[tid] = s;
    tc[tid] = b_r[tid] - m * s;
  }
  for (int e = tid; e < 4 * 67; e += 256) {
    int rr = e / 67, cc = e % 67;
    yr[rr][cc] = y[(size_t)(i0 + rr) * 67 + cc];
  }
  __syncthreads();
  const int o = tid & 63, rs = tid >> 6;
  float acc = 0.f;
  for (int c = 0; c < 67; ++c)
    acc += (yr[rs][c] * sc[c] + tc[c]) * Wr[c * C_OUT + o];
  float v = acc + br[o] + shortcut[(size_t)(i0 + rs) * C_OUT + o];
  dec[(size_t)(i0 + rs) * C_OUT + o]    = v;
  dec_bf[(size_t)(i0 + rs) * C_OUT + o] = f2bf(v);
}

// v = bn(x) (+ add); out = relu(v). Writes bf16 copy; fp32 copy optional.
__global__ void bn_post_kernel(const float* __restrict__ x, const float* __restrict__ stats,
                               const float* __restrict__ g, const float* __restrict__ b,
                               const float* __restrict__ add, float* __restrict__ out_f32,
                               unsigned short* __restrict__ out_bf, int has_add) {
  int e = blockIdx.x * blockDim.x + threadIdx.x;
  if (e >= N_KEEP * C_OUT) return;
  int c = e & 63;
  float m   = stats[c] * (1.f / N_KEEP);
  float var = stats[C_OUT + c] * (1.f / N_KEEP) - m * m;
  float v   = (x[e] - m) * rsqrtf(var + EPS) * g[c] + b[c];
  if (has_add) v += add[e];
  v = v > 0.f ? v : 0.f;
  if (out_f32) out_f32[e] = v;
  out_bf[e] = f2bf(v);
}

// logits[h][n] = dec[n] . Wh[h] + bh[h]
__global__ void logits_kernel(const float* __restrict__ dec, const float* __restrict__ Wh,
                              const float* __restrict__ bh, float* __restrict__ out) {
  int n = blockIdx.x * blockDim.x + threadIdx.x;
  if (n >= N_KEEP) return;
  float a0 = 0.f, a1 = 0.f;
  const float* dr = dec + (size_t)n * C_OUT;
#pragma unroll
  for (int c = 0; c < C_OUT; ++c) {
    float v = dr[c];
    a0 += v * Wh[c];
    a1 += v * Wh[C_OUT + c];
  }
  out[n]          = a0 + bh[0];
  out[N_KEEP + n] = a1 + bh[1];
}

// ---------------------------------------------------------------------------
extern "C" void kernel_launch(void* const* d_in, const int* in_sizes, int n_in,
                              void* d_out, int out_size, void* d_ws, size_t ws_size,
                              hipStream_t stream) {
  (void)in_sizes; (void)n_in; (void)out_size; (void)ws_size;

  const float* x_feats  = (const float*)d_in[0];
  const float* shortcut = (const float*)d_in[1];
  const float* Wup      = (const float*)d_in[2];
  const float* g_up     = (const float*)d_in[3];
  const float* b_up     = (const float*)d_in[4];
  const float* g_r      = (const float*)d_in[5];
  const float* b_r      = (const float*)d_in[6];
  const float* Wr       = (const float*)d_in[7];
  const float* br       = (const float*)d_in[8];
  const float* Wres     = (const float*)d_in[9];
  const float* g_res    = (const float*)d_in[10];
  const float* b_res    = (const float*)d_in[11];
  const float* Wh       = (const float*)d_in[12];
  const float* bh       = (const float*)d_in[13];
  const float* coords_f = (const float*)d_in[14];
  const int*   keep_idx = (const int*)d_in[15];
  const int*   nbr      = (const int*)d_in[16];

  char* ws = (char*)d_ws;
  float*          up      = (float*)(ws + 0);                  // 32,768,000 B
  float*          y       = (float*)(ws + 32768000);           // 26,800,000 B
  float*          tmp     = (float*)(ws + 59568000);           // 25,600,000 B
  unsigned short* dec_bf  = (unsigned short*)(ws + 85168000);  // 100001*64*2 = 12,800,128 B
  unsigned short* h_bf    = (unsigned short*)(ws + 97968128);  // 12,800,128 B
  unsigned short* wupT    = (unsigned short*)(ws + 110768256); //    131,072 B
  unsigned short* wresT   = (unsigned short*)(ws + 110899328); //  1,327,104 B
  float*          statsUp = (float*)(ws + 112226432);          // 128 floats
  float*          statsY  = statsUp + 128;                     // 134 floats
  float*          statsR  = statsY + 134;                      // 128 floats

  float* dec    = (float*)d_out;                  // 100000 x 64
  float* logits = dec + (size_t)N_KEEP * C_OUT;   // 2 x 100000

  const int NW = 8 * C_IN * C_OUT + 6 * 27 * C_OUT * C_OUT;
  cvt_weights_kernel<<<(NW + 255) / 256, 256, 0, stream>>>(Wup, Wres, wupT, wresT);
  zero_pad_kernel<<<1, 64, 0, stream>>>(dec_bf, h_bf);

  gemm_up_kernel<<<dim3(N_COARSE / 16, 4), 256, 0, stream>>>(x_feats, wupT, up);
  col_stats_kernel<<<C_OUT, 256, 0, stream>>>(up, N_FINE, C_OUT, statsUp);

  build_y_kernel<<<(N_KEEP * 67 + 255) / 256, 256, 0, stream>>>(up, statsUp, g_up, b_up,
                                                                keep_idx, coords_f, y);
  col_stats_kernel<<<67, 256, 0, stream>>>(y, N_KEEP, 67, statsY);

  dec0_gemm_kernel<<<N_KEEP / 4, 256, 0, stream>>>(y, statsY, g_r, b_r, Wr, br,
                                                   shortcut, dec, dec_bf);

  for (int i = 0; i < 3; ++i) {
    const unsigned short* w0 = wresT + (size_t)(i * 2 + 0) * 27 * C_OUT * C_OUT;
    const unsigned short* w1 = wresT + (size_t)(i * 2 + 1) * 27 * C_OUT * C_OUT;

    spconv_wmma_kernel<<<N_KEEP / 32, 256, 0, stream>>>(dec_bf, w0, nbr, tmp);
    col_stats_kernel<<<C_OUT, 256, 0, stream>>>(tmp, N_KEEP, C_OUT, statsR);
    bn_post_kernel<<<(N_KEEP * C_OUT + 255) / 256, 256, 0, stream>>>(
        tmp, statsR, g_res + (i * 2 + 0) * C_OUT, b_res + (i * 2 + 0) * C_OUT,
        nullptr, nullptr, h_bf, 0);

    spconv_wmma_kernel<<<N_KEEP / 32, 256, 0, stream>>>(h_bf, w1, nbr, tmp);
    col_stats_kernel<<<C_OUT, 256, 0, stream>>>(tmp, N_KEEP, C_OUT, statsR);
    bn_post_kernel<<<(N_KEEP * C_OUT + 255) / 256, 256, 0, stream>>>(
        tmp, statsR, g_res + (i * 2 + 1) * C_OUT, b_res + (i * 2 + 1) * C_OUT,
        dec, dec, dec_bf, 1);
  }

  logits_kernel<<<(N_KEEP + 255) / 256, 256, 0, stream>>>(dec, Wh, bh, logits);
}
